// MixingBlock_14319420965487
// MI455X (gfx1250) — compile-verified, matrix operand order checked by reading, not run on hardware
//
#include <hip/hip_runtime.h>

#define B_  8
#define N_  2048
#define CS_ 256
#define CV_ 128
#define EPS_ 1e-6f

typedef __attribute__((ext_vector_type(16))) _Float16     v16h;
typedef __attribute__((ext_vector_type(8)))  float        v8f;
typedef __attribute__((ext_vector_type(8)))  unsigned int v8u;
typedef __attribute__((ext_vector_type(4)))  unsigned int v4u;

__device__ __forceinline__ unsigned short f2h(float x) {
    _Float16 h = (_Float16)x;
    return __builtin_bit_cast(unsigned short, h);
}

// A-fragment for v_wmma_f32_16x16x32_f16 from an LDS f16 matrix [16][stride].
// Layout (ISA 7.12.2, 16-bit A 16x32): lane m=lane&15 for both halves;
// vgpr i holds K pair (i<4?0:16) + half*8 + (i&3)*2. Pairs are contiguous,
// so the 8 dwords are two 16B chunks at k0+half*8 and k0+16+half*8.
__device__ __forceinline__ v16h load_a_frag(const unsigned short* base, int stride,
                                            int m, int half, int k0) {
    const unsigned short* p = base + m * stride + k0 + half * 8;
    v4u lo = *(const v4u*)p;
    v4u hi = *(const v4u*)(p + 16);
    v8u u;
    u[0]=lo[0]; u[1]=lo[1]; u[2]=lo[2]; u[3]=lo[3];
    u[4]=hi[0]; u[5]=hi[1]; u[6]=hi[2]; u[7]=hi[3];
    return __builtin_bit_cast(v16h, u);
}

__device__ __forceinline__ v16h load_b_frag(const unsigned int* p) {
    v8u u = *(const v8u*)p;
    return __builtin_bit_cast(v16h, u);
}

__device__ __forceinline__ v8f wmma_f16(v16h a, v16h b, v8f c) {
    return __builtin_amdgcn_wmma_f32_16x16x32_f16(false, a, false, b, (short)0, c, false, false);
}

__device__ __forceinline__ float red16(float v) {
    v += __shfl_xor(v, 1);
    v += __shfl_xor(v, 2);
    v += __shfl_xor(v, 4);
    v += __shfl_xor(v, 8);
    return v;
}

// Pre-swizzle a row-major f32 weight matrix W[P][O] (P = KS*32, O = NT*16)
// into per-lane f16 B-fragments: frag[nt][ks][lane][i] (uints of 2 packed f16).
// B layout mirrors A: lane n = lane&15, vgpr i holds K pair
// (i<4?0:16) + half*8 + (i&3)*2 of column n within the 16-wide tile.
__global__ void prep_frags(const float* __restrict__ W, unsigned int* __restrict__ dst,
                           int O, int KS, int NT) {
    int t = blockIdx.x * blockDim.x + threadIdx.x;
    int total = NT * KS * 256;
    if (t >= total) return;
    int i    = t & 7;
    int lane = (t >> 3) & 31;
    int rest = t >> 8;
    int ks   = rest % KS;
    int nt   = rest / KS;
    int half = lane >> 4, n = lane & 15;
    int kb = ((i & 4) ? 16 : 0) + half * 8 + (i & 3) * 2;
    int k  = ks * 32 + kb;
    int col = nt * 16 + n;
    unsigned short lo = f2h(W[(size_t)k * O + col]);
    unsigned short hi = f2h(W[(size_t)(k + 1) * O + col]);
    dst[t] = (unsigned int)lo | ((unsigned int)hi << 16);
}

__global__ __launch_bounds__(256)
void mixing_block_kernel(const float* __restrict__ x_s,
                         const float* __restrict__ x_v,
                         const float* __restrict__ coords,
                         const unsigned int* __restrict__ wf_cs,
                         const unsigned int* __restrict__ wf_cv,
                         const unsigned int* __restrict__ wf_ss,
                         const unsigned int* __restrict__ wf_sv,
                         float* __restrict__ out) {
    // Static LDS: 20480 + 16384 + 24576 + 576 = 62016 B
    __shared__ __align__(16) unsigned short s_buf[10240];   // f16 staging / LN f16 copies
    __shared__ __align__(16) float s_hs[16 * 256];          // h_s -> s1 -> y_s -> e_s
    __shared__ __align__(16) float s_hv[3 * 16 * 128];      // h_v -> v1 -> y_v -> e_v ([d][m][c])
    __shared__ float s_rh[16 * 3 * 3];                      // rhat [m][k][3]

    const int tid  = threadIdx.x;
    const int wv   = tid >> 5;
    const int lane = tid & 31;
    const int half = lane >> 4;
    const int n16  = lane & 15;
    const int b0   = blockIdx.x >> 7;          // 128 row-tiles per batch
    const int n0   = (blockIdx.x & 127) << 4;

    const float* xs_b = x_s + (size_t)b0 * N_ * CS_;
    const float* xv_b = x_v + (size_t)b0 * N_ * CV_ * 3;
    const float* cd_b = coords + (size_t)b0 * N_ * 3;

    // ---------------- P0: rhat (edge-clamped coords; dn<=1e-6 -> 0) ----------------
    if (tid < 48) {
        int m = tid / 3, k = tid % 3;
        int nc = n0 + m;
        int nb = nc + k - 1;
        nb = nb < 0 ? 0 : (nb > N_ - 1 ? N_ - 1 : nb);
        float dx = cd_b[nb*3+0] - cd_b[nc*3+0];
        float dy = cd_b[nb*3+1] - cd_b[nc*3+1];
        float dz = cd_b[nb*3+2] - cd_b[nc*3+2];
        float dn = sqrtf(dx*dx + dy*dy + dz*dz);
        float s = dn > 1e-6f ? 1.0f / dn : 0.0f;
        s_rh[tid*3+0] = dx*s; s_rh[tid*3+1] = dy*s; s_rh[tid*3+2] = dz*s;
    }
    __syncthreads();

    // ------------- P1: scalar conv, accumulate over taps k (K-dim 1152) -------------
    // scal_in row layout per tap: [ x_s (256) | vdot (128) ]
    v8f acc0 = {0,0,0,0,0,0,0,0};
    v8f acc1 = {0,0,0,0,0,0,0,0};
    for (int k = 0; k < 3; ++k) {
        for (int t = tid; t < 16 * 384; t += 256) {
            int m = t / 384, q = t % 384;
            int nb = n0 + m + k - 1;
            float val = 0.0f;
            if (nb >= 0 && nb < N_) {
                if (q < CS_) {
                    val = xs_b[(size_t)nb * CS_ + q];
                } else {
                    int c = q - CS_;
                    const float* v  = &xv_b[((size_t)nb * CV_ + c) * 3];
                    const float* rh = &s_rh[(m*3 + k) * 3];
                    val = v[0]*rh[0] + v[1]*rh[1] + v[2]*rh[2];
                }
            }
            s_buf[t] = f2h(val);
        }
        __syncthreads();
        const unsigned int* bp0 = wf_cs + ((size_t)wv       * 36 + k*12) * 256 + lane * 8;
        const unsigned int* bp1 = wf_cs + ((size_t)(wv + 8) * 36 + k*12) * 256 + lane * 8;
        #pragma unroll
        for (int ks = 0; ks < 12; ++ks) {
            v16h a = load_a_frag(s_buf, 384, n16, half, ks * 32);
            acc0 = wmma_f16(a, load_b_frag(bp0 + ks * 256), acc0);
            acc1 = wmma_f16(a, load_b_frag(bp1 + ks * 256), acc1);
        }
        __syncthreads();
    }
    {
        const float sc = 0.0294627825f;  // 1/sqrt(1152)
        #pragma unroll
        for (int r = 0; r < 8; ++r) {
            s_hs[(r + 8*half)*256 + wv*16       + n16] = acc0[r] * sc;
            s_hs[(r + 8*half)*256 + (wv+8)*16   + n16] = acc1[r] * sc;
        }
    }

    // --------- P2: vector conv per Cartesian component (K-dim 1536 each) -----------
    // vec_in row layout per tap: [ v_win (128) | s*rhat (256) | v x rhat (128) ]
    for (int d = 0; d < 3; ++d) {
        int c1 = (d + 1) % 3, c2 = (d + 2) % 3;
        v8f acc = {0,0,0,0,0,0,0,0};
        for (int k = 0; k < 3; ++k) {
            for (int t = tid; t < 16 * 512; t += 256) {
                int m = t >> 9, q = t & 511;
                int nb = n0 + m + k - 1;
                float val = 0.0f;
                if (nb >= 0 && nb < N_) {
                    const float* rh = &s_rh[(m*3 + k) * 3];
                    if (q < CV_) {
                        val = xv_b[((size_t)nb * CV_ + q) * 3 + d];
                    } else if (q < CV_ + CS_) {
                        val = xs_b[(size_t)nb * CS_ + (q - CV_)] * rh[d];
                    } else {
                        int c = q - CV_ - CS_;
                        const float* v = &xv_b[((size_t)nb * CV_ + c) * 3];
                        val = v[c1]*rh[c2] - v[c2]*rh[c1];  // (v x rhat)_d
                    }
                }
                s_buf[t] = f2h(val);
            }
            __syncthreads();
            const unsigned int* bp = wf_cv + ((size_t)wv * 48 + k*16) * 256 + lane * 8;
            #pragma unroll
            for (int ks = 0; ks < 16; ++ks) {
                v16h a = load_a_frag(s_buf, 512, n16, half, ks * 32);
                acc = wmma_f16(a, load_b_frag(bp + ks * 256), acc);
            }
            __syncthreads();
        }
        const float sc = 0.0255155137f;  // 1/sqrt(1536)
        #pragma unroll
        for (int r = 0; r < 8; ++r)
            s_hv[(d*16 + r + 8*half)*128 + wv*16 + n16] = acc[r] * sc;
    }
    __syncthreads();

    // ---------------- P3: LN1 (per row: 16 cooperating lanes) ----------------------
    {
        int m = tid >> 4, t = tid & 15;
        float sum = 0.f, sq = 0.f;
        for (int i = 0; i < 16; ++i) {
            float v = s_hs[m*256 + t + 16*i];
            sum += v; sq += v*v;
        }
        sum = red16(sum); sq = red16(sq);
        float mu  = sum * (1.0f/256.0f);
        float inv = rsqrtf(sq * (1.0f/256.0f) - mu*mu + EPS_);
        unsigned short* lns = s_buf;
        for (int i = 0; i < 16; ++i) {
            int c = t + 16*i;
            float v = (s_hs[m*256 + c] - mu) * inv;
            s_hs[m*256 + c] = v;
            lns[m*256 + c] = f2h(v);
        }
        float vn = 0.f;
        for (int i = 0; i < 8; ++i) {
            int c = t + 16*i;
            float a = s_hv[(0*16+m)*128 + c];
            float b = s_hv[(1*16+m)*128 + c];
            float g = s_hv[(2*16+m)*128 + c];
            vn += a*a + b*b + g*g;
        }
        vn = red16(vn);
        float invv = rsqrtf(vn * (1.0f/128.0f) + EPS_);
        unsigned short* lnv = s_buf + 4096;
        for (int i = 0; i < 8; ++i) {
            int c = t + 16*i;
            #pragma unroll
            for (int d = 0; d < 3; ++d) {
                float v = s_hv[(d*16+m)*128 + c] * invv;
                s_hv[(d*16+m)*128 + c] = v;
                lnv[(d*16+m)*128 + c] = f2h(v);
            }
        }
    }
    __syncthreads();

    // -------- P4: SelfInteraction (WMMA, residual add into f32 LDS) ----------------
    {
        const unsigned short* lns = s_buf;
        const unsigned short* lnv = s_buf + 4096;
        for (int jj = 0; jj < 5; ++jj) {
            int j = wv * 5 + jj;  // 40 tile-jobs: 16 scalar + 24 vector
            v8f acc = {0,0,0,0,0,0,0,0};
            if (j < 16) {
                const unsigned int* bp = wf_ss + ((size_t)j * 8) * 256 + lane * 8;
                #pragma unroll
                for (int ks = 0; ks < 8; ++ks) {
                    v16h a = load_a_frag(lns, 256, n16, half, ks * 32);
                    acc = wmma_f16(a, load_b_frag(bp + ks * 256), acc);
                }
                const float sc = 0.0625f;  // 1/sqrt(256)
                #pragma unroll
                for (int r = 0; r < 8; ++r)
                    s_hs[(r + 8*half)*256 + j*16 + n16] += acc[r] * sc;
            } else {
                int d = (j - 16) >> 3, nt = (j - 16) & 7;
                const unsigned int* bp = wf_sv + ((size_t)nt * 4) * 256 + lane * 8;
                #pragma unroll
                for (int ks = 0; ks < 4; ++ks) {
                    v16h a = load_a_frag(lnv + d * 2048, 128, n16, half, ks * 32);
                    acc = wmma_f16(a, load_b_frag(bp + ks * 256), acc);
                }
                const float sc = 0.0883883476f;  // 1/sqrt(128)
                #pragma unroll
                for (int r = 0; r < 8; ++r)
                    s_hv[(d*16 + r + 8*half)*128 + nt*16 + n16] += acc[r] * sc;
            }
        }
    }
    __syncthreads();

    // -------- P5: LN2 -> residual with inputs -> LN3 -> store ----------------------
    {
        int m = tid >> 4, t = tid & 15;
        float sum = 0.f, sq = 0.f;
        for (int i = 0; i < 16; ++i) {
            float v = s_hs[m*256 + t + 16*i];
            sum += v; sq += v*v;
        }
        sum = red16(sum); sq = red16(sq);
        float mu2  = sum * (1.0f/256.0f);
        float inv2 = rsqrtf(sq * (1.0f/256.0f) - mu2*mu2 + EPS_);
        float vn = 0.f;
        for (int i = 0; i < 8; ++i) {
            int c = t + 16*i;
            float a = s_hv[(0*16+m)*128 + c];
            float b = s_hv[(1*16+m)*128 + c];
            float g = s_hv[(2*16+m)*128 + c];
            vn += a*a + b*b + g*g;
        }
        vn = red16(vn);
        float inv2v = rsqrtf(vn * (1.0f/128.0f) + EPS_);

        const float* xsr = xs_b + (size_t)(n0 + m) * CS_;
        const float* xvr = xv_b + (size_t)(n0 + m) * CV_ * 3;
        float sum3 = 0.f, sq3 = 0.f;
        for (int i = 0; i < 16; ++i) {
            int c = t + 16*i;
            float e = xsr[c] + (s_hs[m*256 + c] - mu2) * inv2;
            s_hs[m*256 + c] = e;
            sum3 += e; sq3 += e*e;
        }
        sum3 = red16(sum3); sq3 = red16(sq3);
        float mu3  = sum3 * (1.0f/256.0f);
        float inv3 = rsqrtf(sq3 * (1.0f/256.0f) - mu3*mu3 + EPS_);
        float vn3 = 0.f;
        for (int i = 0; i < 8; ++i) {
            int c = t + 16*i;
            #pragma unroll
            for (int d = 0; d < 3; ++d) {
                float e = xvr[c*3 + d] + s_hv[(d*16+m)*128 + c] * inv2v;
                s_hv[(d*16+m)*128 + c] = e;
                vn3 += e*e;
            }
        }
        vn3 = red16(vn3);
        float inv3v = rsqrtf(vn3 * (1.0f/128.0f) + EPS_);

        float* os = out + (size_t)(b0 * N_ + n0 + m) * CS_;
        for (int i = 0; i < 16; ++i) {
            int c = t + 16*i;
            os[c] = (s_hs[m*256 + c] - mu3) * inv3;
        }
        float* ov = out + (size_t)B_ * N_ * CS_ + (size_t)(b0 * N_ + n0 + m) * CV_ * 3;
        for (int i = 0; i < 8; ++i) {
            int c = t + 16*i;
            #pragma unroll
            for (int d = 0; d < 3; ++d)
                ov[c*3 + d] = s_hv[(d*16+m)*128 + c] * inv3v;
        }
    }
}

extern "C" void kernel_launch(void* const* d_in, const int* in_sizes, int n_in,
                              void* d_out, int out_size, void* d_ws, size_t ws_size,
                              hipStream_t stream) {
    (void)in_sizes; (void)n_in; (void)out_size; (void)ws_size;
    const float* x_s      = (const float*)d_in[0];
    const float* x_v      = (const float*)d_in[1];
    const float* coords   = (const float*)d_in[2];
    const float* W_conv_s = (const float*)d_in[3];
    const float* W_conv_v = (const float*)d_in[4];
    const float* W_si_s   = (const float*)d_in[5];
    const float* W_si_v   = (const float*)d_in[6];

    // Workspace: pre-swizzled f16 weight fragments (packed f16 pairs as u32).
    unsigned int* wf_cs = (unsigned int*)d_ws;        // [16][36][32][8] = 147456 u32
    unsigned int* wf_cv = wf_cs + 16 * 36 * 256;      // [ 8][48][32][8] =  98304 u32
    unsigned int* wf_ss = wf_cv + 8 * 48 * 256;       // [16][ 8][32][8] =  32768 u32
    unsigned int* wf_sv = wf_ss + 16 * 8 * 256;       // [ 8][ 4][32][8] =   8192 u32

    prep_frags<<<(16*36*256 + 255)/256, 256, 0, stream>>>(W_conv_s, wf_cs, 256, 36, 16);
    prep_frags<<<( 8*48*256 + 255)/256, 256, 0, stream>>>(W_conv_v, wf_cv, 128, 48,  8);
    prep_frags<<<(16* 8*256 + 255)/256, 256, 0, stream>>>(W_si_s,   wf_ss, 256,  8, 16);
    prep_frags<<<( 8* 4*256 + 255)/256, 256, 0, stream>>>(W_si_v,   wf_sv, 128,  4,  8);

    mixing_block_kernel<<<B_ * (N_ / 16), 256, 0, stream>>>(
        x_s, x_v, coords, wf_cs, wf_cv, wf_ss, wf_sv, (float*)d_out);
}
